// PointNetPPVariational_10771777979127
// MI455X (gfx1250) — matrix operand order repaired and checked by reading, not compiled, not run on hardware
//
#include <hip/hip_runtime.h>
#include <hip/hip_bf16.h>

#define NB      8
#define NPTS    2048
#define KNN     10
#define NPOINTS (NB * NPTS)          // 16384
#define M_ROWS  (NPOINTS * KNN)      // 163840

typedef __attribute__((ext_vector_type(16))) __bf16        v16bf;
typedef __attribute__((ext_vector_type(8)))  float         v8f;
typedef __attribute__((ext_vector_type(4)))  unsigned int  u32x4;

// ---------- bf16 helpers (round-to-nearest-even), no __bf16 scalar math ----------
__device__ __forceinline__ unsigned short f2bf(float f) {
  unsigned int u = __float_as_uint(f);
  u += 0x7FFFu + ((u >> 16) & 1u);
  return (unsigned short)(u >> 16);
}
__device__ __forceinline__ float bf2f(unsigned short h) {
  return __uint_as_float(((unsigned int)h) << 16);
}

// ---------- KNN (k=10, loop=True -> self at d=0 always selected) ----------
__global__ void knn_kernel(const float* __restrict__ pos, int* __restrict__ nbr) {
  int tid = blockIdx.x * blockDim.x + threadIdx.x;
  if (tid >= NPOINTS) return;
  int b = tid / NPTS, n = tid % NPTS;
  const float* pb = pos + (size_t)b * NPTS * 2;
  float px = pb[n * 2], py = pb[n * 2 + 1];
  float best[KNN]; int bi[KNN];
#pragma unroll
  for (int t = 0; t < KNN; ++t) { best[t] = 3.4e38f; bi[t] = 0; }
  for (int j = 0; j < NPTS; ++j) {
    float dx = pb[j * 2] - px, dy = pb[j * 2 + 1] - py;
    float d = dx * dx + dy * dy;
    int id = j;
#pragma unroll
    for (int t = 0; t < KNN; ++t) {      // register-resident bubble insert
      if (d < best[t]) {
        float tf = best[t]; best[t] = d; d = tf;
        int   ti = bi[t];   bi[t] = id;  id = ti;
      }
    }
  }
#pragma unroll
  for (int t = 0; t < KNN; ++t) nbr[(size_t)tid * KNN + t] = bi[t];
}

// ---------- build message matrix A[M, Kp] = [h_j | pos_j - pos_i | 0-pad] in bf16 ----------
__global__ void build_msg(const float* __restrict__ feat0,
                          const unsigned short* __restrict__ hprev,
                          const float* __restrict__ pos,
                          const int* __restrict__ nbr,
                          unsigned short* __restrict__ A,
                          int Cin, int Kp, int useF32) {
  long tid = (long)blockIdx.x * blockDim.x + threadIdx.x;
  long total = (long)M_ROWS * Kp;
  if (tid >= total) return;
  int  c   = (int)(tid % Kp);
  long row = tid / Kp;
  int  k   = (int)(row % KNN);
  long pt  = row / KNN;                 // b*N + n
  int  b   = (int)(pt / NPTS), n = (int)(pt % NPTS);
  float v = 0.f;
  if (c < Cin) {
    int j = nbr[pt * KNN + k];
    if (useF32) v = feat0[((size_t)b * NPTS + j) * Cin + c];
    else        v = bf2f(hprev[((size_t)b * NPTS + j) * Cin + c]);
  } else if (c < Cin + 2) {
    int j = nbr[pt * KNN + k];
    int d = c - Cin;
    v = pos[((size_t)b * NPTS + j) * 2 + d] - pos[((size_t)b * NPTS + n) * 2 + d];
  }
  A[row * Kp + c] = f2bf(v);
}

// ---------- transpose + zero-pad weights to bf16: Wt[n*Kp + k] = w[k, n] ----------
__global__ void convert_w(const float* __restrict__ w, unsigned short* __restrict__ Wt,
                          int din, int Kp, int dout) {
  int tid = blockIdx.x * blockDim.x + threadIdx.x;
  if (tid >= Kp * dout) return;
  int kk = tid % Kp, n = tid / Kp;
  float v = (kk < din) ? w[(size_t)kk * dout + n] : 0.f;
  Wt[(size_t)n * Kp + kk] = f2bf(v);
}

__global__ void zero_f32(float* __restrict__ p, int n) {
  int t = blockIdx.x * blockDim.x + threadIdx.x;
  if (t < n) p[t] = 0.f;
}

// ---------- WMMA GEMM: Y[M,Nout] = A[M,Kp](bf16) x Wt^T + bias, plus BN partial sums ----
// 256 threads = 8 waves. Block tile = 256(M) x 16(N). Each wave owns two 16x16 C tiles
// (two independent v_wmma dependency chains). The 16-column weight panel is staged into
// LDS once per block with gfx1250 async-to-LDS loads; B fragments then come from
// ds_load_b128 (row stride padded 16B -> conflict-free across the 16 lanes).
__global__ __launch_bounds__(256) void wmma_gemm(
    const unsigned short* __restrict__ A,
    const unsigned short* __restrict__ Wt,
    const float* __restrict__ bias,
    float* __restrict__ Y,
    float* __restrict__ stats,          // [0..Nout) sum, [Nout..2Nout) sumsq
    int Kp, int Nout) {
  __shared__ unsigned short sB[16 * (512 + 8)];   // 16 columns x (Kp+8) bf16, padded
  __shared__ float s_sum[16];
  __shared__ float s_sq[16];

  const int tid  = threadIdx.x;
  const int lane = tid & 31;
  const int wave = tid >> 5;
  const int hh   = lane >> 4;           // K-half select per ISA 16-bit layouts
  const int mr   = lane & 15;
  const int ldb  = Kp + 8;
  const int tileN = blockIdx.x;

  if (tid < 16) { s_sum[tid] = 0.f; s_sq[tid] = 0.f; }

  // ---- async-stage weight panel: Wt rows [tileN*16 .. +16) -> sB (padded) ----
  {
    const int cpr = Kp >> 3;            // 16-byte chunks per row
    const int chunks = 16 * cpr;
    for (int i = tid; i < chunks; i += 256) {
      const int n = i / cpr, cch = i - n * cpr;
      const unsigned short* g = Wt + (size_t)(tileN * 16 + n) * Kp + cch * 8;
      unsigned lds = (unsigned)(size_t)(sB + n * ldb + cch * 8);
      asm volatile("global_load_async_to_lds_b128 %0, %1, off"
                   :: "v"(lds), "v"(g) : "memory");
    }
    asm volatile("s_wait_asynccnt 0x0" ::: "memory");
  }
  __syncthreads();

  const int tileM = (blockIdx.y * 8 + wave) * 2;  // two consecutive 16-row tiles
  const unsigned short* arow0 = A + (size_t)(tileM * 16 + mr) * Kp;
  const unsigned short* arow1 = arow0 + (size_t)16 * Kp;
  const unsigned short* bls   = sB + mr * ldb;    // this lane's column in LDS
  const int col = tileN * 16 + mr;

  v8f acc0, acc1;
  const float bv = bias[col];
#pragma unroll
  for (int j = 0; j < 8; ++j) { acc0[j] = bv; acc1[j] = bv; }

  union Frag { u32x4 u[2]; v16bf v; };
  const int nk = Kp >> 5;
  for (int kc = 0; kc < nk; ++kc) {
    const int kb = kc * 32;
    Frag fa0, fa1, fb;
    // A 16x32 bf16 fragment: lanes 0-15 hold K[0..7]+K[16..23], lanes 16-31 K[8..15]+K[24..31]
    fa0.u[0] = *(const u32x4*)(arow0 + kb + hh * 8);
    fa0.u[1] = *(const u32x4*)(arow0 + kb + 16 + hh * 8);
    fa1.u[0] = *(const u32x4*)(arow1 + kb + hh * 8);
    fa1.u[1] = *(const u32x4*)(arow1 + kb + 16 + hh * 8);
    // B 32x16 bf16 fragment from LDS: lanes 0-15 K[0..7]+K[16..23], lanes 16-31 K[8..15]+K[24..31]
    fb.u[0] = *(const u32x4*)(bls + kb + hh * 8);
    fb.u[1] = *(const u32x4*)(bls + kb + 16 + hh * 8);
    __builtin_prefetch(arow0 + kb + 256, 0, 1);   // global_prefetch_b8 next A panels
    acc0 = __builtin_amdgcn_wmma_f32_16x16x32_bf16(
        false, fa0.v, false, fb.v, (short)0, acc0, false, false);
    acc1 = __builtin_amdgcn_wmma_f32_16x16x32_bf16(
        false, fa1.v, false, fb.v, (short)0, acc1, false, false);
  }

  float lsum = 0.f, lsq = 0.f;
#pragma unroll
  for (int j = 0; j < 8; ++j) {         // D: lane col = mr, rows = hh*8 + j
    float v = acc0[j];
    Y[(size_t)(tileM * 16 + hh * 8 + j) * Nout + col] = v;
    lsum += v; lsq += v * v;
    v = acc1[j];
    Y[(size_t)(tileM * 16 + 16 + hh * 8 + j) * Nout + col] = v;
    lsum += v; lsq += v * v;
  }
  atomicAdd(&s_sum[mr], lsum);          // ds_add_f32 across the 8 waves (same 16 columns)
  atomicAdd(&s_sq[mr], lsq);
  __syncthreads();
  if (tid < 16) {
    int cc = tileN * 16 + tid;
    atomicAdd(&stats[cc], s_sum[tid]);              // global_atomic_add_f32
    atomicAdd(&stats[Nout + cc], s_sq[tid]);
  }
}

// ---------- BN stat finalize: scale = g*rsqrt(var+eps), shift = be - mean*scale ----------
__global__ void bn_finalize(const float* __restrict__ stats, const float* __restrict__ g,
                            const float* __restrict__ be, float* __restrict__ scsh, int Nout) {
  int c = blockIdx.x * blockDim.x + threadIdx.x;
  if (c >= Nout) return;
  const float inv = 1.0f / (float)M_ROWS;
  float mean = stats[c] * inv;
  float var  = stats[Nout + c] * inv - mean * mean;
  float rs   = rsqrtf(var + 1e-5f);
  float sc   = g[c] * rs;
  scsh[c] = sc;
  scsh[Nout + c] = be[c] - mean * sc;
}

// ---------- BN + ReLU + cast bf16 (input panel for GEMM2) ----------
__global__ void bn_relu_cast(const float* __restrict__ Y, const float* __restrict__ scsh,
                             unsigned short* __restrict__ A, int Nout) {
  long tid = (long)blockIdx.x * blockDim.x + threadIdx.x;
  if (tid >= (long)M_ROWS * Nout) return;
  int c = (int)(tid % Nout);
  float v = Y[tid] * scsh[c] + scsh[Nout + c];
  A[tid] = f2bf(fmaxf(v, 0.f));
}

// ---------- BN + max over k=10 messages + ReLU -> h (bf16) ----------
__global__ void bn_max_relu(const float* __restrict__ Y, const float* __restrict__ scsh,
                            unsigned short* __restrict__ h, int Nout) {
  long tid = (long)blockIdx.x * blockDim.x + threadIdx.x;
  if (tid >= (long)NPOINTS * Nout) return;
  int  c  = (int)(tid % Nout);
  long pt = tid / Nout;
  float sc = scsh[c], sh = scsh[Nout + c];
  float m = -3.4e38f;
#pragma unroll
  for (int k = 0; k < KNN; ++k)
    m = fmaxf(m, Y[((size_t)pt * KNN + k) * Nout + c] * sc + sh);
  h[tid] = f2bf(fmaxf(m, 0.f));
}

// ---------- global max pool over N ----------
__global__ void gpool_kernel(const unsigned short* __restrict__ h, float* __restrict__ out, int C) {
  int tid = blockIdx.x * blockDim.x + threadIdx.x;
  if (tid >= NB * C) return;
  int b = tid / C, c = tid % C;
  float m = -3.4e38f;
  for (int n = 0; n < NPTS; ++n)
    m = fmaxf(m, bf2f(h[((size_t)b * NPTS + n) * C + c]));
  out[tid] = m;
}

// ---------- FC head (tiny: 8 rows) ----------
__global__ void fc_kernel(const float* __restrict__ x, const float* __restrict__ w,
                          const float* __restrict__ bias, float* __restrict__ y,
                          int din, int dout, int doRelu, int splitOut) {
  int tid = blockIdx.x * blockDim.x + threadIdx.x;
  if (tid >= NB * dout) return;
  int b = tid / dout, oc = tid % dout;
  float acc = bias[oc];
  for (int i = 0; i < din; ++i) acc += x[(size_t)b * din + i] * w[(size_t)i * dout + oc];
  if (doRelu) acc = fmaxf(acc, 0.f);
  if (splitOut) {                        // (mu, logvar) tuple -> concat flat
    int hC = dout / 2;
    if (oc < hC) y[(size_t)b * hC + oc] = acc;
    else         y[(size_t)NB * hC + (size_t)b * hC + (oc - hC)] = acc;
  } else {
    y[tid] = acc;
  }
}

extern "C" void kernel_launch(void* const* d_in, const int* in_sizes, int n_in,
                              void* d_out, int out_size, void* d_ws, size_t ws_size,
                              hipStream_t stream) {
  (void)in_sizes; (void)n_in; (void)out_size; (void)ws_size;
  const float* features = (const float*)d_in[0];
  const float* pos      = (const float*)d_in[1];
  // d_in[2] = batch (all zeros; batching is the leading dim) -- unused.
  // params pytree order: convs[l] keys sorted (b1,b2,be1,be2,g1,g2,w1,w2), then fc[j] (b,w)
  const int PB = 3;
  auto cp = [&](int l, int w) -> const float* { return (const float*)d_in[PB + l * 8 + w]; };
  const float *fcb[3], *fcw[3];
  for (int j = 0; j < 3; ++j) {
    fcb[j] = (const float*)d_in[PB + 32 + j * 2];
    fcw[j] = (const float*)d_in[PB + 32 + j * 2 + 1];
  }

  // workspace carve-out
  char* ws = (char*)d_ws; size_t off = 0;
  auto wsa = [&](size_t bytes) -> void* {
    void* p = ws + off; off = (off + bytes + 255) & ~(size_t)255; return p;
  };
  int*            nbr   = (int*)wsa((size_t)NPOINTS * KNN * 4);
  unsigned short* Abuf  = (unsigned short*)wsa((size_t)M_ROWS * 512 * 2);
  float*          Ybuf  = (float*)wsa((size_t)M_ROWS * 512 * 4);
  unsigned short* Wt    = (unsigned short*)wsa((size_t)512 * 512 * 2);
  float*          stats = (float*)wsa(2 * 512 * 4);
  float*          scsh  = (float*)wsa(2 * 512 * 4);
  unsigned short* hbuf  = (unsigned short*)wsa((size_t)NPOINTS * 512 * 2);
  float*          gp    = (float*)wsa(NB * 512 * 4);
  float*          t1    = (float*)wsa(NB * 256 * 4);
  float*          t2    = (float*)wsa(NB * 128 * 4);

  knn_kernel<<<(NPOINTS + 127) / 128, 128, 0, stream>>>(pos, nbr);

  const int Cin[4]  = {4, 64, 128, 256};
  const int Dout[4] = {64, 128, 256, 512};
  for (int l = 0; l < 4; ++l) {
    const int cin = Cin[l], dm = cin + 2, dout = Dout[l];
    const int Kp = ((dm + 31) / 32) * 32;              // 32, 96, 160, 288
    const dim3 gg(dout / 16, M_ROWS / 256);

    long tm = (long)M_ROWS * Kp;
    build_msg<<<(unsigned)((tm + 255) / 256), 256, 0, stream>>>(
        features, hbuf, pos, nbr, Abuf, cin, Kp, l == 0 ? 1 : 0);

    // conv1: m @ w1 + b1 -> BN -> ReLU
    convert_w<<<(Kp * dout + 255) / 256, 256, 0, stream>>>(cp(l, 6), Wt, dm, Kp, dout);
    zero_f32<<<(2 * dout + 255) / 256, 256, 0, stream>>>(stats, 2 * dout);
    wmma_gemm<<<gg, 256, 0, stream>>>(Abuf, Wt, cp(l, 0), Ybuf, stats, Kp, dout);
    bn_finalize<<<(dout + 255) / 256, 256, 0, stream>>>(stats, cp(l, 4), cp(l, 2), scsh, dout);
    long tr = (long)M_ROWS * dout;
    bn_relu_cast<<<(unsigned)((tr + 255) / 256), 256, 0, stream>>>(Ybuf, scsh, Abuf, dout);

    // conv2: x @ w2 + b2 -> BN -> max over k -> ReLU
    convert_w<<<(dout * dout + 255) / 256, 256, 0, stream>>>(cp(l, 7), Wt, dout, dout, dout);
    zero_f32<<<(2 * dout + 255) / 256, 256, 0, stream>>>(stats, 2 * dout);
    wmma_gemm<<<gg, 256, 0, stream>>>(Abuf, Wt, cp(l, 1), Ybuf, stats, dout, dout);
    bn_finalize<<<(dout + 255) / 256, 256, 0, stream>>>(stats, cp(l, 5), cp(l, 3), scsh, dout);
    long tp = (long)NPOINTS * dout;
    bn_max_relu<<<(unsigned)((tp + 255) / 256), 256, 0, stream>>>(Ybuf, scsh, hbuf, dout);
  }

  gpool_kernel<<<(NB * 512 + 255) / 256, 256, 0, stream>>>(hbuf, gp, 512);
  fc_kernel<<<(NB * 256 + 255) / 256, 256, 0, stream>>>(gp, fcw[0], fcb[0], t1, 512, 256, 1, 0);
  fc_kernel<<<(NB * 128 + 255) / 256, 256, 0, stream>>>(t1, fcw[1], fcb[1], t2, 256, 128, 1, 0);
  fc_kernel<<<(NB * 256 + 255) / 256, 256, 0, stream>>>(t2, fcw[2], fcb[2], (float*)d_out, 128, 256, 0, 1);
}
